// StateProbabilitiesTimeDepLindblad_22024592293961
// MI455X (gfx1250) — matrix operand order repaired and checked by reading, not compiled, not run on hardware
//
#include <hip/hip_runtime.h>
#include <hip/hip_bf16.h>
#include <math.h>

// ============================================================================
// Lindblad master-equation evolution for MI455X (gfx1250, wave32, WMMA).
//
// Reference shapes (hardcoded): B=256 batch, T=256 steps, D=32, ND=16, NL=16.
//
//   S  = sum_t w_t L_t^dag L_t          (Hermitian, precomputed)
//   A0 = -i*H0 - 0.5*S ; G_d = -i*H_d   (precomputed)
//   A(u) = A0 + sum_d u_d G_d           -> computed as a WMMA GEMM per 8 steps
//   L'_t = sqrt(w_t) L_t                (precomputed bf16 planes)
//   Q = A rho ;  P_t = rho L_t^dag (= (L_t rho)^dag)
//   rho += inv_dt * ( Q + Q^dag + sum_t L'_t P_t )      (all Hermitian terms)
//
// All matmuls: v_wmma_f32_16x16x32_bf16 (bf16 operands, f32 accumulate).
// rho master copy in f32 accumulator registers (one 16x16 quadrant/wave).
// P buffer is double-buffered (ping-pong by chunk parity) so T3 of chunk c+1
// overlaps T4 of chunk c: 5 barriers/step. LDS = 144 KB -> 2 workgroups/WGP.
// ============================================================================

typedef __attribute__((ext_vector_type(16))) __bf16    v16bf;
typedef __attribute__((ext_vector_type(8)))  float     v8f;
typedef __attribute__((ext_vector_type(4)))  unsigned  u32x4;

union Frag {
    u32x4    q[2];   // two 16B chunks (ds_load_b128 each)
    unsigned u[8];   // 8 dwords = 16 packed bf16
    v16bf    v;      // WMMA operand
};

// round-to-nearest-even (prep kernel only)
static __device__ __forceinline__ unsigned short f2bf_rne(float f) {
    unsigned u = __builtin_bit_cast(unsigned, f);
    u += 0x7FFFu + ((u >> 16) & 1u);
    return (unsigned short)(u >> 16);
}
// truncation: bits[31:16] -> folds into ds_store_b16_d16_hi
static __device__ __forceinline__ unsigned short f2bf_hi(float f) {
    return (unsigned short)(__builtin_bit_cast(unsigned, f) >> 16);
}
// pack two bf16 into one dword; HW packed convert (RNE) when available
static __device__ __forceinline__ unsigned pack_bf2(float lo, float hi) {
#if __has_builtin(__builtin_amdgcn_cvt_pk_bf16_f32)
    return __builtin_bit_cast(unsigned, __builtin_amdgcn_cvt_pk_bf16_f32(lo, hi));
#else
    const unsigned ulo = __builtin_bit_cast(unsigned, lo);
    const unsigned uhi = __builtin_bit_cast(unsigned, hi);
    return (ulo >> 16) | (uhi & 0xFFFF0000u);
#endif
}

static __device__ __forceinline__ v8f vzero8() {
    v8f z = {0.f, 0.f, 0.f, 0.f, 0.f, 0.f, 0.f, 0.f};
    return z;
}

static __device__ __forceinline__ v8f wmma_bf16(v16bf a, v16bf b, v8f c) {
    return __builtin_amdgcn_wmma_f32_16x16x32_bf16(
        false, a, false, b, (short)0, c, false, false);
}

// Negate all 16 packed bf16 values (flip sign bits).
static __device__ __forceinline__ Frag fneg(Frag f) {
#pragma unroll
    for (int i = 0; i < 8; ++i) f.u[i] ^= 0x80008000u;
    return f;
}

// A-fragment (16x32, rows Mi..Mi+15) from a row-major bf16 32x32 plane.
static __device__ __forceinline__ Frag load_af(const char* plane, int Mi, int lane) {
    Frag f;
    const char* p = plane + (Mi + (lane & 15)) * 64 + (lane >> 4) * 16;
    f.q[0] = *(const u32x4*)p;
    f.q[1] = *(const u32x4*)(p + 32);
    return f;
}

// B-fragment (32x16, cols Nj..Nj+15): lanes 0-15 K=0..7,16..23, lanes 16-31 K=8..15,24..31,
// lane&15 = column. Reads planes where element (K,col) is at col*64 + K*2:
// col-major planes of O, or row-major planes of conj(O^dag) (dagger sign
// handled in the complex-arithmetic formulas).
static __device__ __forceinline__ Frag load_bf(const char* plane, int Nj, int lane) {
    Frag f;
    const char* p = plane + (Nj + (lane & 15)) * 64 + (lane >> 4) * 16;
    f.q[0] = *(const u32x4*)p;
    f.q[1] = *(const u32x4*)(p + 32);
    return f;
}

// ---------------------------------------------------------------------------
// Prep kernel. ws layout (bytes):
//   [0     ) Lre[16][1024] bf16      32768   (staged into LDS)
//   [32768 ) Lim[16][1024] bf16      32768   (staged into LDS)
//   [65536 ) GfB[128][32][16] bf16  131072   (B-frag-ordered Gflat, L2 use)
//   [196608) Gre[17][1024] bf16      34816   (scratch)
//   [231424) Gim[17][1024] bf16      34816   (scratch)   total 266240 B
//
// Gflat[K][e], e in [0,2048): K<16 -> G_K = -i*H_K ; K==16 -> A0 ; K>16 -> 0.
// e<1024 -> re-plane element e; e>=1024 -> im-plane element e-1024.
// ---------------------------------------------------------------------------
__global__ void __launch_bounds__(1024) prep_ops(
    const float* __restrict__ H0,    // [32][32] complex interleaved
    const float* __restrict__ Hops,  // [16][32][32] complex
    const float* __restrict__ Lb,    // [16][32][32] complex
    const float* __restrict__ w,     // [16]
    char* __restrict__ ws)
{
    const int tid = threadIdx.x;          // 0..1023 -> (i,k)

    {
        const int i = tid >> 5, k = tid & 31;
        float sre = 0.f, sim = 0.f;
        for (int t = 0; t < 16; ++t) {
            const float wt = w[t];
            const float* Lt = Lb + t * 2048;
            float ar = 0.f, ai = 0.f;
            for (int j = 0; j < 32; ++j) {
                const float pr = Lt[(j * 32 + i) * 2], pi = Lt[(j * 32 + i) * 2 + 1];
                const float qr = Lt[(j * 32 + k) * 2], qi = Lt[(j * 32 + k) * 2 + 1];
                ar += pr * qr + pi * qi;
                ai += pr * qi - pi * qr;
            }
            sre += wt * ar;
            sim += wt * ai;
        }

        unsigned short* Lre = (unsigned short*)ws;
        unsigned short* Lim = (unsigned short*)(ws + 32768);
        unsigned short* Gre = (unsigned short*)(ws + 196608);
        unsigned short* Gim = (unsigned short*)(ws + 231424);

        // A0 = -i*H0 - 0.5*S :  -i*(a+bi) = b - a*i
        const float h0r = H0[tid * 2], h0i = H0[tid * 2 + 1];
        Gre[tid] = f2bf_rne(h0i - 0.5f * sre);
        Gim[tid] = f2bf_rne(-h0r - 0.5f * sim);

        for (int d = 0; d < 16; ++d) {
            const float hr = Hops[d * 2048 + tid * 2];
            const float hi = Hops[d * 2048 + tid * 2 + 1];
            Gre[(1 + d) * 1024 + tid] = f2bf_rne(hi);
            Gim[(1 + d) * 1024 + tid] = f2bf_rne(-hr);
        }
        for (int t = 0; t < 16; ++t) {
            const float sw = sqrtf(w[t]);
            Lre[t * 1024 + tid] = f2bf_rne(sw * Lb[t * 2048 + tid * 2]);
            Lim[t * 1024 + tid] = f2bf_rne(sw * Lb[t * 2048 + tid * 2 + 1]);
        }
    }
    __syncthreads();

    // Phase 2: gather Gflat into B-frag order.
    {
        const unsigned short* Gre = (const unsigned short*)(ws + 196608);
        const unsigned short* Gim = (const unsigned short*)(ws + 231424);
        unsigned short* GfB = (unsigned short*)(ws + 65536);
        for (int n = 0; n < 64; ++n) {
            const int o   = n * 1024 + tid;
            const int j   = o >> 9;
            const int rem = o & 511;
            const int ln  = rem >> 4;
            const int kk  = rem & 15;
            const int col = j * 16 + (ln & 15);
            const int K   = (kk < 8) ? 8 * (ln >> 4) + kk : 16 + 8 * (ln >> 4) + (kk - 8);
            unsigned short v = 0;
            if (K <= 16) {
                const int plane = (K < 16) ? (1 + K) : 0;
                v = (col < 1024) ? Gre[plane * 1024 + col]
                                 : Gim[plane * 1024 + (col - 1024)];
            }
            GfB[o] = v;
        }
    }
}

// ---------------------------------------------------------------------------
// Main kernel: one workgroup (4 waves) per batch element.
// Wave w owns output quadrant (Mi, Nj) = ((w&1)*16, (w>>1)*16).
//
// Dynamic LDS (147456 B = 144 KB -> 2 workgroups/WGP):
//   [0     ) Lre planes                        32768
//   [32768 ) Lim planes                        32768
//   [65536 ) P buffer: 8 slots (ping-pong)     32768
//   [98304 ) rho row-major re/im bf16           4096
//   [102400) rho col-major re/im bf16           4096
//   [106496) Q re/im f32 row-major              8192   (reused for output)
//   [114688) Achunk: 8 steps x (re+im bf16)    32768
// ---------------------------------------------------------------------------
__global__ void __launch_bounds__(128) lindblad_evolve(
    const float* __restrict__ x,      // [B][T][16]
    const char*  __restrict__ opsL,   // L planes blob (65536 B)
    const char*  __restrict__ gfb,    // GfB blob (131072 B)
    float* __restrict__ out,          // [B][32]
    int T, float inv_dt)
{
    extern __shared__ char smem[];
    const int tid  = threadIdx.x;
    const int lane = tid & 31;
    const int wave = tid >> 5;
    const int Mi   = (wave & 1) * 16;
    const int Nj   = (wave >> 1) * 16;
    const int b    = blockIdx.x;

    char* Lre    = smem;
    char* Lim    = smem + 32768;
    char* Pbuf   = smem + 65536;       // 8 slots x 4096 B (re 2048 | im 2048)
    char* rRr    = smem + 98304;       // rho row-major
    char* rRi    = smem + 100352;
    char* rCr    = smem + 102400;      // rho col-major
    char* rCi    = smem + 104448;
    char* Qre    = smem + 106496;      // f32 32x32 row-major
    char* Qim    = smem + 110592;
    char* Achunk = smem + 114688;      // step s -> +(s&7)*4096 (re | im)

    // ---- one-time: stage L planes global -> LDS (65536 B) ----
    {
        const u32x4* src = (const u32x4*)opsL;
        u32x4*       dst = (u32x4*)smem;
        for (int i = tid; i < 65536 / 16; i += 128) dst[i] = src[i];
    }
    __syncthreads();

    v8f rho_re = vzero8();
    v8f rho_im = vzero8();
    if (tid == 0) rho_re[0] = 1.0f;          // rho0 = |0><0|

    const int r0 = Mi + 8 * (lane >> 4);     // D-layout row base
    const int cc = Nj + (lane & 15);         // D-layout column

    for (int s = 0; s < T; ++s) {
        // ---- per-8-steps: Achunk = U x Gflat via WMMA (rows 8..15 unused) ----
        if ((s & 7) == 0) {
            Frag Uf;
            {
                const int r = lane & 15, half = lane >> 4;
                int sr = s + (r & 7); if (sr >= T) sr = T - 1;
                const float* xr = x + ((size_t)b * T + sr) * 16 + half * 8;
#pragma unroll
                for (int k = 0; k < 4; ++k)
                    Uf.u[k] = pack_bf2(xr[2 * k], xr[2 * k + 1]);
                Uf.u[4] = (half == 0) ? 0x00003F80u : 0u;  // K=16 slot: coeff 1.0
                Uf.u[5] = 0u; Uf.u[6] = 0u; Uf.u[7] = 0u;
            }
            for (int j = wave * 32; j < wave * 32 + 32; ++j) {
                Frag Bf;
                const char* p = gfb + j * 1024 + lane * 32;
                Bf.q[0] = *(const u32x4*)p;
                Bf.q[1] = *(const u32x4*)(p + 16);
                const v8f d = wmma_bf16(Uf.v, Bf.v, vzero8());
                if (lane < 16) {    // output rows 0..7 = valid steps only
                    char* base = Achunk + (j * 16 + (lane & 15)) * 2;
#pragma unroll
                    for (int v = 0; v < 8; ++v)
                        *(unsigned short*)(base + v * 4096) = f2bf_hi(d[v]);
                }
            }
        }

        // ---- publish rho planes ----
        {
            u32x4 sR, sI;
#pragma unroll
            for (int k = 0; k < 4; ++k) {
                sR[k] = pack_bf2(rho_re[2 * k], rho_re[2 * k + 1]);
                sI[k] = pack_bf2(rho_im[2 * k], rho_im[2 * k + 1]);
            }
            *(u32x4*)(rCr + cc * 64 + r0 * 2) = sR;     // col-major: vector store
            *(u32x4*)(rCi + cc * 64 + r0 * 2) = sI;
#pragma unroll
            for (int v = 0; v < 8; ++v) {               // row-major: d16_hi stores
                const int row = r0 + v;
                *(unsigned short*)(rRr + row * 64 + cc * 2) = f2bf_hi(rho_re[v]);
                *(unsigned short*)(rRi + row * 64 + cc * 2) = f2bf_hi(rho_im[v]);
            }
        }

        if (s + 1 < T) __builtin_prefetch(x + ((size_t)b * T + s + 1) * 16, 0, 0);
        __syncthreads();   // B1: rho planes + Achunk ready

        const char* Are = Achunk + (s & 7) * 4096;
        const char* Aim = Are + 2048;

        const Frag A_r     = load_af(Are, Mi, lane);
        const Frag A_i     = load_af(Aim, Mi, lane);
        const Frag rhoB_r  = load_bf(rCr, Nj, lane);
        const Frag rhoB_i  = load_bf(rCi, Nj, lane);
        const Frag rhoA_r  = load_af(rRr, Mi, lane);
        const Frag rhoA_i  = load_af(rRi, Mi, lane);
        const Frag rhoA_rn = fneg(rhoA_r);

        // Q = A * rho  (qR = Ar*pr - Ai*pi ; qI = Ar*pi + Ai*pr)
        v8f qR = wmma_bf16(A_r.v, rhoB_r.v, vzero8());
        qR     = wmma_bf16(A_i.v, fneg(rhoB_i).v, qR);
        v8f qI = wmma_bf16(A_r.v, rhoB_i.v, vzero8());
        qI     = wmma_bf16(A_i.v, rhoB_r.v, qI);

        // publish Q tile (f32 row-major) for the Q^dag exchange
#pragma unroll
        for (int v = 0; v < 8; ++v) {
            *(float*)(Qre + (r0 + v) * 128 + cc * 4) = qR[v];
            *(float*)(Qim + (r0 + v) * 128 + cc * 4) = qI[v];
        }

        v8f accR = vzero8();
        v8f accI = vzero8();

        // T3: P_t = rho * L_t^dag into ping-pong slot set (ch&1)
        auto t3_chunk = [&](int ch) {
#pragma unroll
            for (int t4 = 0; t4 < 4; ++t4) {
                const int t = ch * 4 + t4;
                const Frag Ld_r = load_bf(Lre + t * 2048, Nj, lane); // L^dag re
                const Frag Ld_i = load_bf(Lim + t * 2048, Nj, lane); // im -> neg
                // pR = pr*Lr + pi*Li ; pI = -pr*Li + pi*Lr
                v8f pR = wmma_bf16(rhoA_r.v,  Ld_r.v, vzero8());
                pR     = wmma_bf16(rhoA_i.v,  Ld_i.v, pR);
                v8f pI = wmma_bf16(rhoA_rn.v, Ld_i.v, vzero8());
                pI     = wmma_bf16(rhoA_i.v,  Ld_r.v, pI);
                u32x4 sR, sI;
#pragma unroll
                for (int k = 0; k < 4; ++k) {
                    sR[k] = pack_bf2(pR[2 * k], pR[2 * k + 1]);
                    sI[k] = pack_bf2(pI[2 * k], pI[2 * k + 1]);
                }
                char* pb = Pbuf + ((ch & 1) * 4 + t4) * 4096;
                *(u32x4*)(pb + cc * 64 + r0 * 2)        = sR;
                *(u32x4*)(pb + 2048 + cc * 64 + r0 * 2) = sI;
            }
        };
        // T4: acc += sum L_t * P_t from slot set (ch&1)
        auto t4_chunk = [&](int ch) {
#pragma unroll
            for (int t4 = 0; t4 < 4; ++t4) {
                const int t = ch * 4 + t4;
                const Frag L_r = load_af(Lre + t * 2048, Mi, lane);
                const Frag L_i = load_af(Lim + t * 2048, Mi, lane);
                const char* pb = Pbuf + ((ch & 1) * 4 + t4) * 4096;
                const Frag P_r = load_bf(pb, Nj, lane);
                const Frag P_i = load_bf(pb + 2048, Nj, lane);
                // re += Lr*pR - Li*pI ; im += Lr*pI + Li*pR
                accR = wmma_bf16(L_r.v, P_r.v,       accR);
                accR = wmma_bf16(L_i.v, fneg(P_i).v, accR);
                accI = wmma_bf16(L_r.v, P_i.v,       accI);
                accI = wmma_bf16(L_i.v, P_r.v,       accI);
            }
        };

        // software-pipelined dissipator: one barrier per chunk
        t3_chunk(0);
        for (int ch = 0; ch < 4; ++ch) {
            __syncthreads();   // P slot set (ch&1) ready; ch==0: Q tiles ready
            if (ch == 0) {
                // acc = Q + Q^dag : element (m,n) += conj(Q[n][m])
                const float* qtR = (const float*)(Qre + cc * 128 + r0 * 4);
                const float* qtI = (const float*)(Qim + cc * 128 + r0 * 4);
#pragma unroll
                for (int v = 0; v < 8; ++v) {
                    accR[v] = qR[v] + qtR[v];
                    accI[v] = qI[v] - qtI[v];
                }
            }
            if (ch < 3) t3_chunk(ch + 1);
            t4_chunk(ch);
        }

        // ---- Euler update in f32 ----
        rho_re += accR * inv_dt;
        rho_im += accI * inv_dt;
    }

    // ---- output |diag(rho)| (reuse Q planes as f32 staging) ----
#pragma unroll
    for (int v = 0; v < 8; ++v) {
        *(float*)(Qre + (r0 + v) * 128 + cc * 4) = rho_re[v];
        *(float*)(Qim + (r0 + v) * 128 + cc * 4) = rho_im[v];
    }
    __syncthreads();
    if (tid < 32) {
        const float re = *(const float*)(Qre + tid * 128 + tid * 4);
        const float im = *(const float*)(Qim + tid * 128 + tid * 4);
        out[(size_t)b * 32 + tid] = sqrtf(re * re + im * im);
    }
}

// ---------------------------------------------------------------------------
extern "C" void kernel_launch(void* const* d_in, const int* in_sizes, int n_in,
                              void* d_out, int out_size, void* d_ws, size_t ws_size,
                              hipStream_t stream) {
    (void)n_in; (void)ws_size;
    const float* x    = (const float*)d_in[0];   // [B,T,16] f32
    const float* H0   = (const float*)d_in[1];   // [32,32] c64
    const float* Hops = (const float*)d_in[2];   // [16,32,32] c64
    const float* w    = (const float*)d_in[3];   // [16] f32
    const float* Lb   = (const float*)d_in[4];   // [16,32,32] c64
    // d_in[5] = timesteps (device scalar); T derived host-side instead:
    const int B = out_size / 32;                 // out is [B,32]
    const int T = in_sizes[0] / (B * 16);        // x is [B,T,16]

    prep_ops<<<1, 1024, 0, stream>>>(H0, Hops, Lb, w, (char*)d_ws);

    const size_t shmem = 147456;                 // 144 KB -> 2 workgroups/WGP
    hipFuncSetAttribute((const void*)lindblad_evolve,
                        hipFuncAttributeMaxDynamicSharedMemorySize, (int)shmem);
    lindblad_evolve<<<B, 128, shmem, stream>>>(
        x, (const char*)d_ws, (const char*)d_ws + 65536,
        (float*)d_out, T, 1.0f / (float)T);
}
